// LayerNormGRU_5050881540668
// MI455X (gfx1250) — matrix operand order repaired and checked by reading, not compiled, not run on hardware
//
#include <hip/hip_runtime.h>
#include <hip/hip_bf16.h>

// ---------------- problem constants ----------------
constexpr int Bsz = 64, Ssz = 512, Isz = 512, Hsz = 1024;
constexpr int G = 3 * Hsz;                    // 3072 gate width
constexpr size_t NROWS = (size_t)Bsz * Ssz;   // 32768

// ---------------- workspace layout -----------------
constexpr size_t align256(size_t x) { return (x + 255) & ~(size_t)255; }
constexpr size_t OFF_BAR  = 0;                                                  // 2 ints
constexpr size_t OFF_WIH  = 256;                                                // ushort[G*I]
constexpr size_t OFF_WHH  = align256(OFF_WIH + sizeof(unsigned short) * (size_t)G * Isz);
constexpr size_t OFF_X    = align256(OFF_WHH + sizeof(unsigned short) * (size_t)G * Hsz);
constexpr size_t OFF_IH   = align256(OFF_X   + sizeof(unsigned short) * NROWS * Isz);
constexpr size_t OFF_HH   = align256(OFF_IH  + sizeof(float) * NROWS * (size_t)G);
constexpr size_t OFF_HF   = align256(OFF_HH  + sizeof(float) * (size_t)Bsz * G);
constexpr size_t OFF_HB   = align256(OFF_HF  + sizeof(float) * (size_t)Bsz * Hsz);
constexpr size_t OFF_MEAN = align256(OFF_HB  + sizeof(unsigned short) * (size_t)Bsz * Hsz);

// ---------------- WMMA types ------------------------
typedef __attribute__((ext_vector_type(16))) __bf16       v16bf;
typedef __attribute__((ext_vector_type(8)))  float        v8f;
typedef __attribute__((ext_vector_type(4)))  unsigned int v4u;

union ABVec { v16bf v; v4u q[2]; };

__device__ inline v4u ld128(const unsigned short* p) {
  return *reinterpret_cast<const v4u*>(p);
}

__device__ inline unsigned short f2bf(float f) {   // RNE f32 -> bf16
  unsigned int u = __float_as_uint(f);
  unsigned int r = u + 0x7FFFu + ((u >> 16) & 1u);
  return (unsigned short)(r >> 16);
}

// Fragment loaders per the ISA 16-bit A(16x32)/B(32x16) lane layouts.
// A: lane L (m = L&15): halves 0-7 -> K = k0 + (L>>4)*8 + e,
//    halves 8-15        -> K = k0 + 16 + (L>>4)*8 + e.
// B: lane L (n = L&15): same K mapping as A.
__device__ inline v16bf loadA(const unsigned short* aRow, int k0, int hi) {
  ABVec a;
  a.q[0] = ld128(aRow + k0 + hi * 8);
  a.q[1] = ld128(aRow + k0 + 16 + hi * 8);
  return a.v;
}
__device__ inline v16bf loadB(const unsigned short* bRow, int k0, int hi) {
  ABVec b;
  b.q[0] = ld128(bRow + k0 + hi * 8);
  b.q[1] = ld128(bRow + k0 + 16 + hi * 8);
  return b.v;
}

__device__ inline v8f wmma_bf16(v16bf a, v16bf b, v8f c) {
  return __builtin_amdgcn_wmma_f32_16x16x32_bf16(false, a, false, b, (short)0, c,
                                                 false, false);
}

// Single 16x16 tile over full K (used by the recurrent scan).
__device__ inline v8f gemm_tile(const unsigned short* __restrict__ aRow,
                                const unsigned short* __restrict__ bRow,
                                int K, int lane) {
  v8f acc = {};
  const int hi = lane >> 4;
  for (int k0 = 0; k0 < K; k0 += 32)
    acc = wmma_bf16(loadA(aRow, k0, hi), loadB(bRow, k0, hi), acc);
  return acc;
}

// ---------------- grid-wide barrier -----------------
__device__ inline void grid_sync(int* bar) {
  __syncthreads();
  if (threadIdx.x == 0) {
    int* cnt = bar;
    int* gen = bar + 1;
    int g = __hip_atomic_load(gen, __ATOMIC_ACQUIRE, __HIP_MEMORY_SCOPE_AGENT);
    __threadfence();
    int prev = __hip_atomic_fetch_add(cnt, 1, __ATOMIC_ACQ_REL, __HIP_MEMORY_SCOPE_AGENT);
    if (prev == (int)gridDim.x - 1) {
      __hip_atomic_store(cnt, 0, __ATOMIC_RELAXED, __HIP_MEMORY_SCOPE_AGENT);
      __hip_atomic_fetch_add(gen, 1, __ATOMIC_ACQ_REL, __HIP_MEMORY_SCOPE_AGENT);
    } else {
      while (__hip_atomic_load(gen, __ATOMIC_ACQUIRE, __HIP_MEMORY_SCOPE_AGENT) == g)
        __builtin_amdgcn_s_sleep(1);
    }
  }
  __syncthreads();
}

// ---------------- kernels ---------------------------
__global__ void __launch_bounds__(256) cvt_bf16_kernel(const float* __restrict__ in,
                                                       unsigned short* __restrict__ out,
                                                       int n) {
  for (int i = blockIdx.x * 256 + threadIdx.x; i < n; i += gridDim.x * 256)
    out[i] = f2bf(in[i]);
}

// ih = X(32768x512) * Wih^T -> IH (32768x3072).
// 2x2 register blocking: one wave -> 32x32 output patch (4 accumulators),
// A fragments reused across 2 N-tiles, B across 2 M-tiles (2 loads/WMMA).
__global__ void __launch_bounds__(256) gemm_ih_kernel(const unsigned short* __restrict__ X,
                                                      const unsigned short* __restrict__ W,
                                                      float* __restrict__ IH) {
  const int lane = threadIdx.x & 31;
  const int wave = threadIdx.x >> 5;
  const int hi = lane >> 4;
  constexpr int NT2 = G / 32;            // 96 patches along N
  const long wi = (long)blockIdx.x * 8 + wave;
  const int mt2 = (int)(wi / NT2);       // 0..1023
  const int nt2 = (int)(wi % NT2);       // 0..95

  const unsigned short* aRow0 = X + (size_t)(mt2 * 32 + (lane & 15)) * Isz;
  const unsigned short* aRow1 = aRow0 + (size_t)16 * Isz;
  const unsigned short* bRow0 = W + (size_t)(nt2 * 32 + (lane & 15)) * Isz;
  const unsigned short* bRow1 = bRow0 + (size_t)16 * Isz;

  v8f acc00 = {}, acc01 = {}, acc10 = {}, acc11 = {};
  for (int k0 = 0; k0 < Isz; k0 += 32) {
    v16bf a0 = loadA(aRow0, k0, hi);
    v16bf a1 = loadA(aRow1, k0, hi);
    v16bf b0 = loadB(bRow0, k0, hi);
    v16bf b1 = loadB(bRow1, k0, hi);
    acc00 = wmma_bf16(a0, b0, acc00);
    acc01 = wmma_bf16(a0, b1, acc01);
    acc10 = wmma_bf16(a1, b0, acc10);
    acc11 = wmma_bf16(a1, b1, acc11);
  }

  const int n0 = nt2 * 32 + (lane & 15);
  const int m0 = mt2 * 32 + hi * 8;
#pragma unroll
  for (int r = 0; r < 8; ++r) {
    IH[(size_t)(m0 + r) * G + n0]           = acc00[r];
    IH[(size_t)(m0 + r) * G + n0 + 16]      = acc01[r];
    IH[(size_t)(m0 + 16 + r) * G + n0]      = acc10[r];
    IH[(size_t)(m0 + 16 + r) * G + n0 + 16] = acc11[r];
  }
}

// Per-row LayerNorm (ddof=1) on [0,2H) and [2H,3H) chunks, in place.
__global__ void __launch_bounds__(256) ln_ih_kernel(float* __restrict__ IH,
                                                    const float* __restrict__ gamma,
                                                    const float* __restrict__ bias) {
  __shared__ float s1[256], s2[256];
  float* p = IH + (size_t)blockIdx.x * G;
  const int tid = threadIdx.x;

  float a = 0.f, b = 0.f;
  for (int j = tid; j < 2 * Hsz; j += 256) { float v = p[j]; a += v; b += v * v; }
  s1[tid] = a; s2[tid] = b; __syncthreads();
  for (int st = 128; st > 0; st >>= 1) {
    if (tid < st) { s1[tid] += s1[tid + st]; s2[tid] += s2[tid + st]; }
    __syncthreads();
  }
  const float nRZ = 2.f * Hsz;
  float meanRZ = s1[0] / nRZ;
  float invRZ  = rsqrtf((s2[0] - s1[0] * s1[0] / nRZ) / (nRZ - 1.f));
  __syncthreads();

  a = 0.f; b = 0.f;
  for (int j = tid; j < Hsz; j += 256) { float v = p[2 * Hsz + j]; a += v; b += v * v; }
  s1[tid] = a; s2[tid] = b; __syncthreads();
  for (int st = 128; st > 0; st >>= 1) {
    if (tid < st) { s1[tid] += s1[tid + st]; s2[tid] += s2[tid + st]; }
    __syncthreads();
  }
  const float nN = (float)Hsz;
  float meanN = s1[0] / nN;
  float invN  = rsqrtf((s2[0] - s1[0] * s1[0] / nN) / (nN - 1.f));
  __syncthreads();

  for (int j = tid; j < G; j += 256) {
    float v = p[j];
    float m  = (j < 2 * Hsz) ? meanRZ : meanN;
    float is = (j < 2 * Hsz) ? invRZ  : invN;
    p[j] = gamma[j] * ((v - m) * is) + bias[j];
  }
}

// Persistent recurrent scan: 64 blocks x 256 threads, grid barrier per phase.
__global__ void __launch_bounds__(256) scan_kernel(const float* __restrict__ IH,
                                                   const unsigned short* __restrict__ WHH,
                                                   float* __restrict__ HH,
                                                   float* __restrict__ hF,
                                                   unsigned short* __restrict__ hB,
                                                   float* __restrict__ means,
                                                   const float* __restrict__ gammaHH,
                                                   const float* __restrict__ biasHH,
                                                   float* __restrict__ out,
                                                   int* bar) {
  const int lane = threadIdx.x & 31;
  const int wave = threadIdx.x >> 5;
  const int gw = blockIdx.x * 8 + wave;   // 0..511
  const int nWaves = gridDim.x * 8;       // 512
  __shared__ float s1[256], s2[256];
  constexpr int TILES = (Bsz / 16) * (G / 16);  // 4*192 = 768

  for (int t = 0; t < Ssz; ++t) {
    // Phase 1: HH = h(64x1024) * WHH^T -> (64x3072), WMMA
    for (int tile = gw; tile < TILES; tile += nWaves) {
      const int mt = tile / (G / 16), nt = tile % (G / 16);
      const unsigned short* aRow = hB  + (size_t)(mt * 16 + (lane & 15)) * Hsz;
      const unsigned short* bRow = WHH + (size_t)(nt * 16 + (lane & 15)) * Hsz;
      v8f acc = gemm_tile(aRow, bRow, Hsz, lane);
      const int n = nt * 16 + (lane & 15);
      const int mbase = mt * 16 + (lane >> 4) * 8;
#pragma unroll
      for (int r = 0; r < 8; ++r)
        HH[(size_t)(mbase + r) * G + n] = acc[r];
    }
    grid_sync(bar);

    // Phase 2: per-batch-row means (block b reduces row b; grid = 64)
    {
      const float* row = HH + (size_t)blockIdx.x * G;
      float a = 0.f, c = 0.f;
      for (int j = threadIdx.x; j < G; j += 256) {
        float v = row[j];
        if (j < 2 * Hsz) a += v; else c += v;
      }
      s1[threadIdx.x] = a; s2[threadIdx.x] = c; __syncthreads();
      for (int st = 128; st > 0; st >>= 1) {
        if (threadIdx.x < st) { s1[threadIdx.x] += s1[threadIdx.x + st];
                                s2[threadIdx.x] += s2[threadIdx.x + st]; }
        __syncthreads();
      }
      if (threadIdx.x == 0) {
        means[blockIdx.x * 2 + 0] = s1[0] / (2.f * Hsz);
        means[blockIdx.x * 2 + 1] = s2[0] / (float)Hsz;
      }
      __syncthreads();
    }
    grid_sync(bar);

    // Phase 3: mean-only LN on hh + GRU gating, write outs + new h
    for (int e = blockIdx.x * 256 + threadIdx.x; e < Bsz * Hsz; e += gridDim.x * 256) {
      const int b = e >> 10;
      const int j = e & (Hsz - 1);
      const float mRZ = means[b * 2 + 0];
      const float mN  = means[b * 2 + 1];
      const float* ihrow = IH + ((size_t)b * Ssz + t) * G;
      const float* hhrow = HH + (size_t)b * G;
      float hr = gammaHH[j]           * (hhrow[j]           - mRZ) + biasHH[j];
      float hz = gammaHH[Hsz + j]     * (hhrow[Hsz + j]     - mRZ) + biasHH[Hsz + j];
      float hn = gammaHH[2 * Hsz + j] * (hhrow[2 * Hsz + j] - mN ) + biasHH[2 * Hsz + j];
      float r = 1.f / (1.f + __expf(-(ihrow[j] + hr)));
      float z = 1.f / (1.f + __expf(-(ihrow[Hsz + j] + hz)));
      float n = tanhf(ihrow[2 * Hsz + j] + r * hn);
      float hNew = (1.f - z) * n + z * hF[e];
      hF[e] = hNew;
      hB[e] = f2bf(hNew);
      out[((size_t)b * Ssz + t) * Hsz + j] = hNew;
      if (t == Ssz - 1)
        out[(size_t)Bsz * Ssz * Hsz + e] = hNew;   // final_state
    }
    grid_sync(bar);
  }
}

// ---------------- launch ----------------------------
extern "C" void kernel_launch(void* const* d_in, const int* in_sizes, int n_in,
                              void* d_out, int out_size, void* d_ws, size_t ws_size,
                              hipStream_t stream) {
  const float* x     = (const float*)d_in[0];
  const float* state = (const float*)d_in[1];
  const float* wih   = (const float*)d_in[2];
  const float* whh   = (const float*)d_in[3];
  const float* bih   = (const float*)d_in[4];
  const float* bhh   = (const float*)d_in[5];
  const float* gih   = (const float*)d_in[6];
  const float* ghh   = (const float*)d_in[7];
  float* out = (float*)d_out;
  char* ws = (char*)d_ws;

  int*            bar  = (int*)(ws + OFF_BAR);
  unsigned short* wihB = (unsigned short*)(ws + OFF_WIH);
  unsigned short* whhB = (unsigned short*)(ws + OFF_WHH);
  unsigned short* xB   = (unsigned short*)(ws + OFF_X);
  float*          IH   = (float*)(ws + OFF_IH);
  float*          HH   = (float*)(ws + OFF_HH);
  float*          hF   = (float*)(ws + OFF_HF);
  unsigned short* hB   = (unsigned short*)(ws + OFF_HB);
  float*          mean = (float*)(ws + OFF_MEAN);

  hipMemsetAsync(bar, 0, 256, stream);

  cvt_bf16_kernel<<<2048, 256, 0, stream>>>(wih,   wihB, G * Isz);
  cvt_bf16_kernel<<<2048, 256, 0, stream>>>(whh,   whhB, G * Hsz);
  cvt_bf16_kernel<<<4096, 256, 0, stream>>>(x,     xB,   (int)(NROWS * Isz));
  cvt_bf16_kernel<<<256,  256, 0, stream>>>(state, hB,   Bsz * Hsz);
  hipMemcpyAsync(hF, state, sizeof(float) * Bsz * Hsz, hipMemcpyDeviceToDevice, stream);

  // 1024 x 96 patches of 32x32, 8 waves/block -> 12288 blocks
  gemm_ih_kernel<<<12288, 256, 0, stream>>>(xB, wihB, IH);
  ln_ih_kernel<<<(int)NROWS, 256, 0, stream>>>(IH, gih, bih);
  scan_kernel<<<64, 256, 0, stream>>>(IH, whhB, HH, hF, hB, mean, ghh, bhh, out, bar);
}
